// CNNCacheModel_31164282700281
// MI455X (gfx1250) — matrix operand order repaired and checked by reading, not compiled, not run on hardware
//
#include <hip/hip_runtime.h>
#include <hip/hip_fp16.h>
#include <math.h>

typedef __attribute__((ext_vector_type(16))) _Float16 v16h;
typedef __attribute__((ext_vector_type(8)))  _Float16 v8h;
typedef __attribute__((ext_vector_type(8)))  float    v8f;

#define Bsz   8
#define Ssz   4096
#define Dsz   1024
#define Lnum  4
#define Ktap  3
#define DBsz  512
#define Nrows (Bsz * Ssz)   // 32768

// GEMM block geometry (shared by both instantiations)
constexpr int BM  = 64;        // rows per block
constexpr int BN  = 256;       // cols per block
constexpr int KC  = 64;        // k-chunk staged in LDS
constexpr int LDT = KC + 8;    // padded LDS row (halves)

__device__ __forceinline__ float gelu_exact(float v) {
    return 0.5f * v * (1.0f + erff(v * 0.70710678118654752f));
}

// ---------------------------------------------------------------- LN stats
__global__ __launch_bounds__(128) void ln_stats_kernel(
        const float* __restrict__ x, float* __restrict__ stats) {
    int wave = threadIdx.x >> 5;
    int lane = threadIdx.x & 31;
    int row  = blockIdx.x * 4 + wave;
    const float* xr = x + (size_t)row * Dsz;
    float s = 0.f, sq = 0.f;
#pragma unroll
    for (int j = 0; j < Dsz / 32; ++j) {
        float v = xr[lane + j * 32];
        s += v; sq += v * v;
    }
#pragma unroll
    for (int m = 16; m >= 1; m >>= 1) {
        s  += __shfl_xor(s,  m, 32);
        sq += __shfl_xor(sq, m, 32);
    }
    if (lane == 0) {
        float mean = s * (1.0f / Dsz);
        float var  = sq * (1.0f / Dsz) - mean * mean;
        stats[row * 2 + 0] = mean;
        stats[row * 2 + 1] = rsqrtf(var + 1e-5f);
    }
}

// ------------------------------------------- LN + causal dilated DW conv + GELU
__global__ __launch_bounds__(256) void ln_conv_gelu_kernel(
        const float* __restrict__ x, const float* __restrict__ stats,
        const float* __restrict__ g, const float* __restrict__ bln,
        const float* __restrict__ w, const float* __restrict__ bconv,
        _Float16* __restrict__ h16, int dil) {
    size_t idx = (size_t)blockIdx.x * blockDim.x + threadIdx.x;
    int d    = (int)(idx & (Dsz - 1));
    int row  = (int)(idx >> 10);
    int s    = row & (Ssz - 1);
    int base = row - s;            // b * S
    float gd = g[d], bd = bln[d];
    float acc = bconv[d];
#pragma unroll
    for (int k = 0; k < Ktap; ++k) {
        int ss = s - (Ktap - 1 - k) * dil;
        if (ss >= 0) {
            int r2 = base + ss;
            float xv   = x[(size_t)r2 * Dsz + d];
            float mean = stats[r2 * 2 + 0];
            float rstd = stats[r2 * 2 + 1];
            float ln   = (xv - mean) * rstd * gd + bd;
            acc += w[d * Ktap + k] * ln;
        }
    }
    h16[idx] = (_Float16)gelu_exact(acc);
}

// ---------------------------------------------------------------- f32 -> f16
__global__ __launch_bounds__(256) void f32_to_f16_kernel(
        const float* __restrict__ src, _Float16* __restrict__ dst, int n) {
    int i = blockIdx.x * blockDim.x + threadIdx.x;
    if (i < n) dst[i] = (_Float16)src[i];
}

// ----------------------------------------------- LDS WMMA fragment loads
// A-matrix 16x32 f16: lanes 0-15 hold K=[0..15],
// lanes 16-31 hold K=[16..31] for row M=lane&15 (matches B).
__device__ __forceinline__ v16h lds_a_frag(const _Float16* base, int kk, int lane) {
    int r = lane & 15, h = lane >> 4;
    const _Float16* p = base + r * LDT + kk + h * 16;
    v8h lo = *(const v8h*)p;
    v8h hi = *(const v8h*)(p + 8);
    return __builtin_shufflevector(lo, hi, 0, 1, 2, 3, 4, 5, 6, 7,
                                           8, 9, 10, 11, 12, 13, 14, 15);
}
// B-matrix 32x16 f16: lane holds column (lane&15); lanes 0-15 K=[0..15],
// lanes 16-31 K=[16..31] -> 16 contiguous halves from the weight row.
__device__ __forceinline__ v16h lds_b_frag(const _Float16* base, int kk, int lane) {
    int c = lane & 15, h = lane >> 4;
    const _Float16* p = base + c * LDT + kk + h * 16;
    v8h lo = *(const v8h*)p;
    v8h hi = *(const v8h*)(p + 8);
    return __builtin_shufflevector(lo, hi, 0, 1, 2, 3, 4, 5, 6, 7,
                                           8, 9, 10, 11, 12, 13, 14, 15);
}

// Register staging buffer: one k-chunk's worth of global data per thread.
struct StageRegs {
    v8h a[2];   // A chunk: 64x64 halves = 512 b128 units / 256 threads
    v8h b[8];   // B chunk: 256x64 halves = 2048 units / 256 threads
};

template<int KDIM>
__device__ __forceinline__ void load_stage(StageRegs& sr,
        const _Float16* __restrict__ Ag, const _Float16* __restrict__ Wg,
        int m0, int n0, int k0, int tid) {
#pragma unroll
    for (int j = 0; j < 2; ++j) {
        int u = tid + j * 256, r = u >> 3, cc = u & 7;
        sr.a[j] = *(const v8h*)(Ag + (size_t)(m0 + r) * KDIM + k0 + cc * 8);
    }
#pragma unroll
    for (int j = 0; j < 8; ++j) {
        int u = tid + j * 256, r = u >> 3, cc = u & 7;
        sr.b[j] = *(const v8h*)(Wg + (size_t)(n0 + r) * KDIM + k0 + cc * 8);
    }
}

__device__ __forceinline__ void store_stage(const StageRegs& sr,
        _Float16* As, _Float16* Bs, int tid) {
#pragma unroll
    for (int j = 0; j < 2; ++j) {
        int u = tid + j * 256, r = u >> 3, cc = u & 7;
        *(v8h*)&As[r * LDT + cc * 8] = sr.a[j];
    }
#pragma unroll
    for (int j = 0; j < 8; ++j) {
        int u = tid + j * 256, r = u >> 3, cc = u & 7;
        *(v8h*)&Bs[r * LDT + cc * 8] = sr.b[j];
    }
}

// --------------------------------------------------- LDS-blocked NT GEMM
// Out[m,n] = sum_k A[m,k] * W[n,k]  (+ bias; GELU->f16  OR  +residual->f32)
// 256 threads = 8 waves (2x4); tile 64x256; k staged in 64-chunks.
// Explicit ping-pong register prefetch + fully unrolled chunk loop so the
// next chunk's global loads retire under the current chunk's WMMA phase
// (no cross-iteration register copy -> no full loadcnt drain).
template<int KDIM, int OUTLD, bool DOGELU>
__global__ __launch_bounds__(256) void gemm_block_kernel(
        const _Float16* __restrict__ Ag, const _Float16* __restrict__ Wg,
        const float* __restrict__ bias, _Float16* __restrict__ out16,
        float* __restrict__ out32, const float* __restrict__ resid) {
    __shared__ alignas(16) _Float16 As[BM * LDT];   //  9 KB
    __shared__ alignas(16) _Float16 Bs[BN * LDT];   // 36 KB
    int tid  = threadIdx.x;
    int wave = tid >> 5, lane = tid & 31;
    int wm = wave & 1, wn = wave >> 1;              // 2 x 4 wave grid
    int m0 = blockIdx.x * BM, n0 = blockIdx.y * BN;

    constexpr int NCHUNK = KDIM / KC;
    v8f acc[2][4] = {};
    StageRegs buf[2];
    load_stage<KDIM>(buf[0], Ag, Wg, m0, n0, 0, tid);

#pragma unroll
    for (int ch = 0; ch < NCHUNK; ++ch) {
        store_stage(buf[ch & 1], As, Bs, tid);
        __syncthreads();

        // Prefetch next chunk into the other register buffer; its loadcnt
        // is only waited at the next chunk's store_stage.
        if (ch + 1 < NCHUNK)
            load_stage<KDIM>(buf[(ch + 1) & 1], Ag, Wg, m0, n0, (ch + 1) * KC, tid);

#pragma unroll
        for (int kk = 0; kk < KC; kk += 32) {
            // Hoist all fragment loads ahead of the WMMA burst.
            v16h a0 = lds_a_frag(&As[(wm * 32     ) * LDT], kk, lane);
            v16h a1 = lds_a_frag(&As[(wm * 32 + 16) * LDT], kk, lane);
            v16h bf[4];
#pragma unroll
            for (int t = 0; t < 4; ++t)
                bf[t] = lds_b_frag(&Bs[(wn * 64 + t * 16) * LDT], kk, lane);
#pragma unroll
            for (int t = 0; t < 4; ++t) {
                acc[0][t] = __builtin_amdgcn_wmma_f32_16x16x32_f16(
                    false, a0, false, bf[t], (short)0, acc[0][t], false, false);
                acc[1][t] = __builtin_amdgcn_wmma_f32_16x16x32_f16(
                    false, a1, false, bf[t], (short)0, acc[1][t], false, false);
            }
        }
        __syncthreads();
    }

    // Epilogue per ISA C/D layout: lane<16 -> N=lane, M=vgpr; lane>=16 -> M=8+vgpr.
    int half = lane >> 4, c = lane & 15;
#pragma unroll
    for (int rt = 0; rt < 2; ++rt)
#pragma unroll
        for (int t = 0; t < 4; ++t) {
            int col = n0 + wn * 64 + t * 16 + c;
            float bc = bias[col];
#pragma unroll
            for (int i = 0; i < 8; ++i) {
                int row = m0 + wm * 32 + rt * 16 + half * 8 + i;
                float v = acc[rt][t][i] + bc;
                if constexpr (DOGELU) {
                    out16[(size_t)row * OUTLD + col] = (_Float16)gelu_exact(v);
                } else {
                    size_t o = (size_t)row * OUTLD + col;
                    out32[o] = v + resid[o];
                }
            }
        }
}

// ------------------------------------------------------------------- driver
extern "C" void kernel_launch(void* const* d_in, const int* in_sizes, int n_in,
                              void* d_out, int out_size, void* d_ws, size_t ws_size,
                              hipStream_t stream) {
    (void)in_sizes; (void)n_in; (void)out_size; (void)ws_size;

    const float* x0       = (const float*)d_in[0];
    const float* ln_scale = (const float*)d_in[1];
    const float* ln_bias  = (const float*)d_in[2];
    const float* dw_w     = (const float*)d_in[3];
    const float* dw_b     = (const float*)d_in[4];
    const float* comp_w   = (const float*)d_in[5];
    const float* comp_b   = (const float*)d_in[6];
    const float* exp_w    = (const float*)d_in[7];
    const float* exp_b    = (const float*)d_in[8];
    float* xout = (float*)d_out;

    // workspace layout
    char* ws = (char*)d_ws;
    _Float16* h16   = (_Float16*)ws;                               // 64 MB
    _Float16* hc16  = (_Float16*)(ws + (size_t)Nrows * Dsz * 2);   // 32 MB
    _Float16* wc16  = (_Float16*)((char*)hc16 + (size_t)Nrows * DBsz * 2); // 1 MB
    _Float16* we16  = (_Float16*)((char*)wc16 + (size_t)DBsz * Dsz * 2);   // 1 MB
    float*    stats = (float*)((char*)we16 + (size_t)Dsz * DBsz * 2);      // 256 KB

    for (int i = 0; i < Lnum; ++i) {
        int dil = 1 << i;
        const float* xcur = (i == 0) ? x0 : (const float*)d_out;

        ln_stats_kernel<<<dim3(Nrows / 4), dim3(128), 0, stream>>>(xcur, stats);

        ln_conv_gelu_kernel<<<dim3((unsigned)((size_t)Nrows * Dsz / 256)),
                              dim3(256), 0, stream>>>(
            xcur, stats, ln_scale + (size_t)i * Dsz, ln_bias + (size_t)i * Dsz,
            dw_w + (size_t)i * Dsz * Ktap, dw_b + (size_t)i * Dsz, h16, dil);

        f32_to_f16_kernel<<<dim3((DBsz * Dsz + 255) / 256), dim3(256), 0, stream>>>(
            comp_w + (size_t)i * DBsz * Dsz, wc16, DBsz * Dsz);
        f32_to_f16_kernel<<<dim3((Dsz * DBsz + 255) / 256), dim3(256), 0, stream>>>(
            exp_w + (size_t)i * Dsz * DBsz, we16, Dsz * DBsz);

        // compress: [32768 x 1024] x [512 x 1024]^T -> gelu -> f16 [32768 x 512]
        gemm_block_kernel<Dsz, DBsz, true>
            <<<dim3(Nrows / 64, DBsz / 256), dim3(256), 0, stream>>>(
                h16, wc16, comp_b + (size_t)i * DBsz, hc16, nullptr, nullptr);

        // expand: [32768 x 512] x [1024 x 512]^T + bias + residual -> f32
        gemm_block_kernel<DBsz, Dsz, false>
            <<<dim3(Nrows / 64, Dsz / 256), dim3(256), 0, stream>>>(
                hc16, we16, exp_b + (size_t)i * Dsz, nullptr, xout, xcur);
    }
}
